// GATGraphClassifier_40802189312729
// MI455X (gfx1250) — matrix operand order repaired and checked by reading, not compiled, not run on hardware
//
#include <hip/hip_runtime.h>
#include <math.h>

#define N_NODES   100000
#define N_EDGES   1600000
#define G_GRAPHS  512
#define C_IN      64
#define HID       32
#define HEADS     4
#define OUT_C     2
#define BN_EPS    1e-5f
#define NEG_SLOPE 0.2f
#define ET        (N_EDGES + N_NODES)   // edges + self loops

typedef __bf16 bhalf;
typedef __attribute__((ext_vector_type(16))) bhalf        v16bf;
typedef __attribute__((ext_vector_type(8)))  float        v8f;
typedef __attribute__((ext_vector_type(8)))  unsigned int v8u;
typedef __attribute__((ext_vector_type(4)))  unsigned int v4u;
typedef __attribute__((ext_vector_type(8)))  int          v8i;
typedef __attribute__((ext_vector_type(4)))  int          v4i;

// ---------------------------------------------------------------- helpers

__device__ __forceinline__ unsigned short f2bf(float f) {
  unsigned int u = __float_as_uint(f);
  unsigned int r = 0x7FFFu + ((u >> 16) & 1u);   // round-to-nearest-even
  return (unsigned short)((u + r) >> 16);
}

__device__ __forceinline__ float atomicMaxF(float* addr, float val) {
  unsigned int* ua = reinterpret_cast<unsigned int*>(addr);
  unsigned int old = *((volatile unsigned int*)ua);
  while (__uint_as_float(old) < val) {
    unsigned int assumed = old;
    old = atomicCAS(ua, assumed, __float_as_uint(val));
    if (old == assumed) break;
  }
  return __uint_as_float(old);
}

__device__ __forceinline__ void edge_sd(int e, const int* __restrict__ ei,
                                        int& s, int& d) {
  if (e < N_EDGES) { s = ei[e]; d = ei[N_EDGES + e]; }
  else             { s = d = e - N_EDGES; }          // self loop
}

// ---------------------------------------------------------------- kernels

__global__ void gat_f32_to_bf16(const float* __restrict__ src,
                                unsigned short* __restrict__ dst, int n) {
  int i = blockIdx.x * blockDim.x + threadIdx.x;
  if (i < n) dst[i] = f2bf(src[i]);
}

__global__ void gat_fill_f32(float* __restrict__ p, float v, int n) {
  int i = blockIdx.x * blockDim.x + threadIdx.x;
  if (i < n) p[i] = v;
}

// Pre-pack weight matrix W[K, NT*16] (f32) into per-lane WMMA B-fragment
// layout: Bp[((nt*(K/32)+kc)*32 + lane)*8 + r] = bf16 pair for
// kb = kc*32 + (lane>=16)*16 + 2r, col = nt*16 + (lane&15).
__global__ void gat_pack_B(const float* __restrict__ W,
                           unsigned int* __restrict__ Bp, int K, int NT) {
  int idx = blockIdx.x * blockDim.x + threadIdx.x;
  int total = NT * (K / 32) * 32 * 8;
  if (idx >= total) return;
  int r    = idx & 7;
  int lane = (idx >> 3) & 31;
  int rest = idx >> 8;
  int kc   = rest % (K / 32);
  int nt   = rest / (K / 32);
  int col  = nt * 16 + (lane & 15);
  int kb   = kc * 32 + (lane >> 4) * 16 + (r << 1);
  int ldb  = NT * 16;
  unsigned int lo = f2bf(W[(size_t)kb * ldb + col]);
  unsigned int hi = f2bf(W[(size_t)(kb + 1) * ldb + col]);
  Bp[idx] = lo | (hi << 16);
}

// GEMM: C[16*gridDim.x, 16*NT] = A[.,K](bf16 row-major) * B(pre-packed)
// A tile staged into LDS via the Tensor Data Mover (TDM), one DMA per block.
template <int K, int NT>
__global__ __launch_bounds__(32 * NT)
void gat_wmma_gemm_bf16(const unsigned short* __restrict__ A,
                        const unsigned int* __restrict__ Bp,
                        float* __restrict__ C) {
  __shared__ unsigned int sA[8 * K];                // 16 rows x K bf16
  const int tid = threadIdx.x;
  const int m0  = blockIdx.x * 16;

  if (tid < 32) {   // wave 0 issues the tensor DMA (EXEC ignored by TDM)
    unsigned long long gaddr =
        (unsigned long long)(uintptr_t)(A + (size_t)m0 * K);
    unsigned int ldsoff = (unsigned int)(uintptr_t)sA;  // low 32b = LDS offset
    // D# group 0: count=1 | lds_addr | global_addr[56:0] | type=2
    v4u g0 = {1u, ldsoff, (unsigned int)gaddr,
              (unsigned int)((gaddr >> 32) & 0x01FFFFFFu) | (2u << 30)};
    // D# group 1: 2D tensor, data_size=2B, dims K x 16, tile K x 16, stride K
    const unsigned int td0 = K, td1 = 16, tile0 = K, tile1 = 16;
    const unsigned long long s0 = K;
    v8u t;
    t[0] = (1u << 16);                                   // data_size = 2 bytes
    t[1] = (td0 & 0xFFFFu) << 16;
    t[2] = (td0 >> 16) | ((td1 & 0xFFFFu) << 16);
    t[3] = (td1 >> 16) | ((tile0 & 0xFFFFu) << 16);
    t[4] = (tile1 & 0xFFFFu);                            // tile_dim2 = 0
    t[5] = (unsigned int)(s0 & 0xFFFFFFFFull);
    t[6] = (unsigned int)((s0 >> 32) & 0xFFFFull);       // stride1 = 0
    t[7] = 0u;
    v8i g1 = __builtin_bit_cast(v8i, t);
    v4i z4 = {0, 0, 0, 0};
    v8i z8 = {0, 0, 0, 0, 0, 0, 0, 0};
    __builtin_amdgcn_tensor_load_to_lds(g0, g1, z4, z4, z8, 0);
    __builtin_amdgcn_s_wait_tensorcnt(0);
  }
  __syncthreads();

  const int lane = tid & 31;
  const int wave = tid >> 5;
  const int half = lane >> 4;     // lane group 0..15 / 16..31
  const int lq   = lane & 15;
  const int n0   = wave * 16;
  const int ldc  = 16 * NT;

  const uint4* sArow = reinterpret_cast<const uint4*>(sA) + lq * (K / 8);
  v8f acc = {};
#pragma unroll
  for (int kk = 0; kk < K; kk += 32) {
    // A fragment (contiguous k = 16*half + s, matching packed B): two contiguous 16B LDS reads
    const int a0i = (kk >> 3) + half * 2;
    uint4 a0 = sArow[a0i];
    uint4 a1 = sArow[a0i + 1];
    v4u a0v = __builtin_bit_cast(v4u, a0);
    v4u a1v = __builtin_bit_cast(v4u, a1);
    v8u au  = __builtin_shufflevector(a0v, a1v, 0, 1, 2, 3, 4, 5, 6, 7);
    // B fragment: pre-packed, one 32B contiguous load per lane
    const size_t boff = (((size_t)wave * (K / 32) + (kk >> 5)) * 32 + lane) * 8;
    v8u bu = *reinterpret_cast<const v8u*>(Bp + boff);
    v16bf av = __builtin_bit_cast(v16bf, au);
    v16bf bv = __builtin_bit_cast(v16bf, bu);
    acc = __builtin_amdgcn_wmma_f32_16x16x32_bf16(
        false, av, false, bv, (short)0, acc, false, false);
  }
  // D layout: VGPR v, lanes<16 -> M=v, lanes>=16 -> M=v+8 ; N = lane&15
  float* Crow = C + (size_t)m0 * ldc + n0 + lq;
#pragma unroll
  for (int v = 0; v < 8; ++v)
    Crow[(size_t)(v + half * 8) * ldc] = acc[v];
}

// per (node, head): a_src / a_dst attention logits
__global__ void gat_att_scores(const float* __restrict__ xs,
                               const float* __restrict__ att_s,
                               const float* __restrict__ att_d,
                               float* __restrict__ as_, float* __restrict__ ad_,
                               int H) {
  int idx = blockIdx.x * blockDim.x + threadIdx.x;
  if (idx >= N_NODES * H) return;
  int n = idx / H, h = idx - n * H;
  const float* row = xs + (size_t)n * H * HID + h * HID;
  float s = 0.f, d = 0.f;
#pragma unroll
  for (int c = 0; c < HID; ++c) {
    s += row[c] * att_s[h * HID + c];
    d += row[c] * att_d[h * HID + c];
  }
  as_[idx] = s;
  ad_[idx] = d;
}

__global__ void gat_edge_max(const int* __restrict__ ei,
                             const float* __restrict__ as_,
                             const float* __restrict__ ad_,
                             float* __restrict__ mx, int H) {
  int idx = blockIdx.x * blockDim.x + threadIdx.x;
  if (idx >= ET * H) return;
  int e = idx / H, h = idx - e * H;
  int s, d; edge_sd(e, ei, s, d);
  float el = as_[s * H + h] + ad_[d * H + h];
  el = el > 0.f ? el : el * NEG_SLOPE;
  atomicMaxF(&mx[d * H + h], el);
}

__global__ void gat_edge_den(const int* __restrict__ ei,
                             const float* __restrict__ as_,
                             const float* __restrict__ ad_,
                             const float* __restrict__ mx,
                             float* __restrict__ den, int H) {
  int idx = blockIdx.x * blockDim.x + threadIdx.x;
  if (idx >= ET * H) return;
  int e = idx / H, h = idx - e * H;
  int s, d; edge_sd(e, ei, s, d);
  float el = as_[s * H + h] + ad_[d * H + h];
  el = el > 0.f ? el : el * NEG_SLOPE;
  atomicAdd(&den[d * H + h], __expf(el - mx[d * H + h]));
}

__global__ void gat_edge_accum(const int* __restrict__ ei,
                               const float* __restrict__ as_,
                               const float* __restrict__ ad_,
                               const float* __restrict__ mx,
                               const float* __restrict__ den,
                               const float* __restrict__ xs,
                               float* __restrict__ out, int H) {
  int idx = blockIdx.x * blockDim.x + threadIdx.x;
  if (idx >= ET * H) return;
  int e = idx / H, h = idx - e * H;
  int s, d; edge_sd(e, ei, s, d);
  const float* srow = xs + (size_t)s * H * HID + h * HID;
  __builtin_prefetch(srow, 0, 0);
  float el = as_[s * H + h] + ad_[d * H + h];
  el = el > 0.f ? el : el * NEG_SLOPE;
  int dh = d * H + h;
  float alpha = __expf(el - mx[dh]) / den[dh];
  float* drow = out + (size_t)d * H * HID + h * HID;
#pragma unroll
  for (int c = 0; c < HID; ++c) atomicAdd(&drow[c], alpha * srow[c]);
}

// +bias -> BN(eval) -> ELU, in place; optional bf16 mirror for next GEMM
__global__ void gat_bn_elu(float* __restrict__ acc,
                           const float* __restrict__ b,
                           const float* __restrict__ gamma,
                           const float* __restrict__ beta,
                           const float* __restrict__ rm,
                           const float* __restrict__ rv,
                           unsigned short* __restrict__ out_bf, int W) {
  int idx = blockIdx.x * blockDim.x + threadIdx.x;
  if (idx >= N_NODES * W) return;
  int j = idx % W;
  float v = acc[idx] + b[j];
  v = (v - rm[j]) * rsqrtf(rv[j] + BN_EPS) * gamma[j] + beta[j];
  v = v > 0.f ? v : (__expf(v) - 1.f);      // ELU
  acc[idx] = v;
  if (out_bf) out_bf[idx] = f2bf(v);
}

__global__ void gat_pool_cnt(const int* __restrict__ batch,
                             float* __restrict__ cnt) {
  int n = blockIdx.x * blockDim.x + threadIdx.x;
  if (n < N_NODES) atomicAdd(&cnt[batch[n]], 1.0f);
}

__global__ void gat_pool_sum(const int* __restrict__ batch,
                             const float* __restrict__ h,
                             float* __restrict__ sums) {
  int idx = blockIdx.x * blockDim.x + threadIdx.x;
  if (idx >= N_NODES * HID) return;
  int n = idx / HID, c = idx - n * HID;
  atomicAdd(&sums[(size_t)batch[n] * HID + c], h[idx]);
}

__global__ void gat_classify(const float* __restrict__ sums,
                             const float* __restrict__ cnt,
                             const float* __restrict__ Wc,
                             const float* __restrict__ bc,
                             float* __restrict__ out) {
  int idx = blockIdx.x * blockDim.x + threadIdx.x;
  if (idx >= G_GRAPHS * OUT_C) return;
  int g = idx / OUT_C, o = idx - g * OUT_C;
  float inv = 1.0f / fmaxf(cnt[g], 1.0f);
  float r = bc[o];
#pragma unroll
  for (int c = 0; c < HID; ++c)
    r += (sums[(size_t)g * HID + c] * inv) * Wc[c * OUT_C + o];
  out[idx] = r;
}

// ---------------------------------------------------------------- launch

static inline unsigned cdiv(long long a, long long b) {
  return (unsigned)((a + b - 1) / b);
}

extern "C" void kernel_launch(void* const* d_in, const int* in_sizes, int n_in,
                              void* d_out, int out_size, void* d_ws, size_t ws_size,
                              hipStream_t stream) {
  (void)in_sizes; (void)n_in; (void)out_size; (void)ws_size;
  const float* x        = (const float*)d_in[0];
  const int*   ei       = (const int*)  d_in[1];
  const int*   batch    = (const int*)  d_in[2];
  const float* W1       = (const float*)d_in[3];
  const float* att_src1 = (const float*)d_in[4];
  const float* att_dst1 = (const float*)d_in[5];
  const float* b1       = (const float*)d_in[6];
  const float* gamma1   = (const float*)d_in[7];
  const float* beta1    = (const float*)d_in[8];
  const float* rm1      = (const float*)d_in[9];
  const float* rv1      = (const float*)d_in[10];
  const float* W2       = (const float*)d_in[11];
  const float* att_src2 = (const float*)d_in[12];
  const float* att_dst2 = (const float*)d_in[13];
  const float* b2       = (const float*)d_in[14];
  const float* gamma2   = (const float*)d_in[15];
  const float* beta2    = (const float*)d_in[16];
  const float* rm2      = (const float*)d_in[17];
  const float* rv2      = (const float*)d_in[18];
  const float* Wc       = (const float*)d_in[19];
  const float* bc       = (const float*)d_in[20];
  float* out = (float*)d_out;

  // ---- workspace layout (256B aligned, with deliberate aliasing) ----
  char* base = (char*)d_ws;
  size_t o = 0;
  auto alloc = [&](size_t bytes) {
    size_t r = o; o += (bytes + 255) & ~(size_t)255; return r;
  };
  const size_t o_xs1  = alloc((size_t)N_NODES * 128 * 4); // layer1 feats; later: h1 bf16
  const size_t o_acc1 = alloc((size_t)N_NODES * 128 * 4); // l1 accum -> h1; later l2 accum -> h2
  const size_t o_xbf  = alloc((size_t)N_NODES * 64 * 2);  // x bf16; later: xs2 fp32
  const size_t o_w1p  = alloc((size_t)8 * 2 * 32 * 8 * 4);  // packed W1 B-frags
  const size_t o_w2p  = alloc((size_t)2 * 4 * 32 * 8 * 4);  // packed W2 B-frags
  const size_t o_as1  = alloc((size_t)N_NODES * HEADS * 4);
  const size_t o_ad1  = alloc((size_t)N_NODES * HEADS * 4);
  const size_t o_mx1  = alloc((size_t)N_NODES * HEADS * 4);
  const size_t o_dn1  = alloc((size_t)N_NODES * HEADS * 4);
  const size_t o_as2  = alloc((size_t)N_NODES * 4);
  const size_t o_ad2  = alloc((size_t)N_NODES * 4);
  const size_t o_mx2  = alloc((size_t)N_NODES * 4);
  const size_t o_dn2  = alloc((size_t)N_NODES * 4);
  const size_t o_sum  = alloc((size_t)G_GRAPHS * HID * 4);
  const size_t o_cnt  = alloc((size_t)G_GRAPHS * 4);

  unsigned short* xbf = (unsigned short*)(base + o_xbf);
  unsigned int*   w1p = (unsigned int*)(base + o_w1p);
  unsigned int*   w2p = (unsigned int*)(base + o_w2p);
  float* xs1  = (float*)(base + o_xs1);
  float* acc1 = (float*)(base + o_acc1);
  unsigned short* h1b = (unsigned short*)(base + o_xs1);   // alias (after edge accum 1)
  float* xs2  = (float*)(base + o_xbf);                    // alias (after GEMM1)
  float* acc2 = (float*)(base + o_acc1);                   // alias (after epilogue 1)
  float* as1 = (float*)(base + o_as1); float* ad1 = (float*)(base + o_ad1);
  float* mx1 = (float*)(base + o_mx1); float* dn1 = (float*)(base + o_dn1);
  float* as2 = (float*)(base + o_as2); float* ad2 = (float*)(base + o_ad2);
  float* mx2 = (float*)(base + o_mx2); float* dn2 = (float*)(base + o_dn2);
  float* sums = (float*)(base + o_sum); float* cnt = (float*)(base + o_cnt);

  const int B = 256;

  // ---- precision conversion + weight packing ----
  gat_f32_to_bf16<<<cdiv((long long)N_NODES * 64, B), B, 0, stream>>>(x, xbf, N_NODES * 64);
  gat_pack_B<<<cdiv(8 * 2 * 256, B), B, 0, stream>>>(W1, w1p, 64, 8);
  gat_pack_B<<<cdiv(2 * 4 * 256, B), B, 0, stream>>>(W2, w2p, 128, 2);

  // ---- layer 1: xs1 = x @ W1 (WMMA bf16, TDM-staged A) ----
  gat_wmma_gemm_bf16<64, 8><<<N_NODES / 16, 256, 0, stream>>>(xbf, w1p, xs1);
  gat_att_scores<<<cdiv(N_NODES * HEADS, B), B, 0, stream>>>(xs1, att_src1, att_dst1, as1, ad1, HEADS);

  gat_fill_f32<<<cdiv(N_NODES * HEADS, B), B, 0, stream>>>(mx1, -INFINITY, N_NODES * HEADS);
  (void)hipMemsetAsync(dn1, 0, (size_t)N_NODES * HEADS * 4, stream);
  gat_edge_max<<<cdiv((long long)ET * HEADS, B), B, 0, stream>>>(ei, as1, ad1, mx1, HEADS);
  gat_edge_den<<<cdiv((long long)ET * HEADS, B), B, 0, stream>>>(ei, as1, ad1, mx1, dn1, HEADS);
  (void)hipMemsetAsync(acc1, 0, (size_t)N_NODES * 128 * 4, stream);
  gat_edge_accum<<<cdiv((long long)ET * HEADS, B), B, 0, stream>>>(ei, as1, ad1, mx1, dn1, xs1, acc1, HEADS);
  gat_bn_elu<<<cdiv((long long)N_NODES * 128, B), B, 0, stream>>>(acc1, b1, gamma1, beta1, rm1, rv1, h1b, 128);

  // ---- layer 2: xs2 = h1 @ W2 (WMMA bf16, TDM-staged A) ----
  gat_wmma_gemm_bf16<128, 2><<<N_NODES / 16, 64, 0, stream>>>(h1b, w2p, xs2);
  gat_att_scores<<<cdiv(N_NODES, B), B, 0, stream>>>(xs2, att_src2, att_dst2, as2, ad2, 1);

  gat_fill_f32<<<cdiv(N_NODES, B), B, 0, stream>>>(mx2, -INFINITY, N_NODES);
  (void)hipMemsetAsync(dn2, 0, (size_t)N_NODES * 4, stream);
  gat_edge_max<<<cdiv((long long)ET, B), B, 0, stream>>>(ei, as2, ad2, mx2, 1);
  gat_edge_den<<<cdiv((long long)ET, B), B, 0, stream>>>(ei, as2, ad2, mx2, dn2, 1);
  (void)hipMemsetAsync(acc2, 0, (size_t)N_NODES * HID * 4, stream);
  gat_edge_accum<<<cdiv((long long)ET, B), B, 0, stream>>>(ei, as2, ad2, mx2, dn2, xs2, acc2, 1);
  gat_bn_elu<<<cdiv((long long)N_NODES * HID, B), B, 0, stream>>>(acc2, b2, gamma2, beta2, rm2, rv2, nullptr, HID);

  // ---- mean pool + classifier ----
  (void)hipMemsetAsync(sums, 0, (size_t)G_GRAPHS * HID * 4, stream);
  (void)hipMemsetAsync(cnt, 0, (size_t)G_GRAPHS * 4, stream);
  gat_pool_cnt<<<cdiv(N_NODES, B), B, 0, stream>>>(batch, cnt);
  gat_pool_sum<<<cdiv((long long)N_NODES * HID, B), B, 0, stream>>>(batch, acc2, sums);
  gat_classify<<<cdiv(G_GRAPHS * OUT_C, B), B, 0, stream>>>(sums, cnt, Wc, bc, out);
}
